// CARAFE_64433099375152
// MI455X (gfx1250) — compile-verified, matrix operand order checked by reading, not run on hardware
//
#include <hip/hip_runtime.h>
#include <hip/hip_bf16.h>

// ---------------------------------------------------------------------------
// CARAFE upsampling for MI455X (gfx1250, wave32, WMMA).
//
// Roofline: ~42 MB unavoidable HBM traffic (~1.8us @ 23.3 TB/s); all
// intermediates are L2-resident (192 MB L2).  The 3x3 encoder conv
// (~1.9 GFLOP) runs on v_wmma_f32_16x16x32_bf16; the reassembly stage is
// store-bandwidth-bound and stays on VALU with the 25-tap sum folded to 9.
//
// Round-2/3 changes:
//  * LDS layouts transposed + padded (272B / 80B row pitch) so every WMMA
//    fragment is built from two conflict-free ds_load_b128 ops instead of
//    16x ds_load_u16 + v_mov_b16 merges.
//  * Kernel-3 x-halo staging uses GLOBAL_LOAD_ASYNC_TO_LDS_B32 (ASYNCcnt)
//    memory->LDS with no VGPR round trip.  The builtin takes int* operands
//    (AS1 global source, AS3 LDS destination).
// ---------------------------------------------------------------------------

typedef __attribute__((ext_vector_type(16))) __bf16 v16bf;
typedef __attribute__((ext_vector_type(8)))  __bf16 v8bf;
typedef __attribute__((ext_vector_type(8)))  float  v8f;

#define NB   4      // batch
#define CIN  128    // input channels
#define HH   64
#define WW   64
#define HW   (HH * WW)        // 4096
#define CC   64     // compressed channels
#define KC   100    // encoder output channels = (2*5)^2
#define KD   (CC * 9)         // im2col K dim = 576
#define HU   (HH * 2)
#define WU   (WW * 2)

// Padded LDS row pitches (in bf16 elements).  Byte pitch must be a multiple
// of 16 (aligned b128 loads) and not a multiple of 256 (bank rotation):
//   136 elem = 272 B = 68 DWORDs -> 4-bank rotation per row, conflict-free.
//   40  elem =  80 B = 20 DWORDs -> 20-bank rotation per row, conflict-free.
#define PITCH_K128  136
#define PITCH_K32   40

#if defined(__gfx1250__) && __has_builtin(__builtin_amdgcn_global_load_async_to_lds_b32)
#define HAVE_ASYNC_LDS 1
#define GPTR(p) ((__attribute__((address_space(1))) int*)(uintptr_t)(p))
#define LPTR(p) ((__attribute__((address_space(3))) int*)(uintptr_t)(p))
#else
#define HAVE_ASYNC_LDS 0
#endif

__device__ __forceinline__ void wait_async_zero() {
#if HAVE_ASYNC_LDS
#if __has_builtin(__builtin_amdgcn_s_wait_asynccnt)
    __builtin_amdgcn_s_wait_asynccnt(0);
#else
    asm volatile("s_wait_asynccnt 0" ::: "memory");
#endif
#endif
}

// Build a 16x32 bf16 WMMA fragment for this lane.  `base` points at the
// lane's row at K=0 of the current 32-wide K step (16B aligned).  Per ISA
// 7.12.2 the lane (half h = lane>>4) holds K runs [8h,8h+8) and [16+8h,+8):
// two contiguous 16-byte chunks -> two ds_load_b128.
__device__ __forceinline__ v16bf load_frag16(const __bf16* base, int half) {
    const v8bf lo = *(const v8bf*)(base + 8 * half);
    const v8bf hi = *(const v8bf*)(base + 16 + 8 * half);
    return __builtin_shufflevector(lo, hi, 0, 1, 2, 3, 4, 5, 6, 7,
                                           8, 9, 10, 11, 12, 13, 14, 15);
}

// ---------------------------------------------------------------------------
// Kernel 1: 1x1 compression conv as GEMM  mid[64,4096] = Wc[64,128] x X[128,4096]
// One block = one batch n, one 64-column strip.  4 waves; each wave owns a
// 16-column panel and all four 16-row M tiles.  K = 128 -> 4 WMMA steps.
// Both operands staged K-contiguous with padded pitch -> b128 fragment loads.
// ---------------------------------------------------------------------------
__global__ __launch_bounds__(128) void carafe_compress_wmma(
    const float* __restrict__ x, const float* __restrict__ comp_w,
    const float* __restrict__ comp_b, __bf16* __restrict__ mid)
{
    __shared__ __align__(16) __bf16 Wc[CC * PITCH_K128];  // [row][k] ~17 KB
    __shared__ __align__(16) __bf16 Xs[64 * PITCH_K128];  // [col][k] ~17 KB

    const int tid = threadIdx.x;
    const int n   = blockIdx.x >> 6;          // 64 strips per batch
    const int p0  = (blockIdx.x & 63) * 64;   // strip start in 0..4095

    for (int i = tid; i < CC * CIN; i += 128) {
        const int row = i >> 7, k = i & 127;
        Wc[row * PITCH_K128 + k] = (__bf16)comp_w[i];
    }
    const float* xb = x + (size_t)n * CIN * HW;
    for (int i = tid; i < CIN * 64; i += 128) {
        const int k = i >> 6, col = i & 63;           // consecutive tid -> col
        Xs[col * PITCH_K128 + k] = (__bf16)xb[k * HW + p0 + col];
    }
    __syncthreads();

    const int lane = tid & 31, wave = tid >> 5;
    const int col  = lane & 15, half = lane >> 4;

    v8f acc[4] = {};
    #pragma unroll
    for (int kk = 0; kk < 4; ++kk) {            // K steps of 32
        const v16bf bfrag =
            load_frag16(&Xs[(wave * 16 + col) * PITCH_K128 + kk * 32], half);
        #pragma unroll
        for (int m = 0; m < 4; ++m) {
            const v16bf afrag =
                load_frag16(&Wc[(m * 16 + col) * PITCH_K128 + kk * 32], half);
            acc[m] = __builtin_amdgcn_wmma_f32_16x16x32_bf16(
                false, afrag, false, bfrag, (short)0, acc[m], false, false);
        }
    }

    // C/D layout: VGPR r -> M = mtile*16 + r + 8*half, N = lane&15.
    #pragma unroll
    for (int m = 0; m < 4; ++m) {
        #pragma unroll
        for (int r = 0; r < 8; ++r) {
            const int row = m * 16 + r + 8 * half;
            const int p   = p0 + wave * 16 + col;
            mid[((size_t)n * CC + row) * HW + p] =
                (__bf16)(acc[m][r] + comp_b[row]);
        }
    }
}

// ---------------------------------------------------------------------------
// Kernel 2: 3x3 encoder conv as im2col GEMM
//   enc[100,4096] = We[100,576] x im2col(mid)[576,4096]
// One block = one batch n, one image row y (64 columns).  The 3-row mid halo
// lives in LDS as the im2col source; a [112,32] bf16 weight slice (padded
// pitch, b128 A-fragment loads) is restaged per K step.  18 K steps x 7 M
// tiles of WMMA per wave.  The B gather stays scalar: each im2col element is
// consumed by exactly one lane, so this is the minimum-LDS-traffic form.
// ---------------------------------------------------------------------------
__global__ __launch_bounds__(128) void carafe_encode_wmma(
    const __bf16* __restrict__ mid, const float* __restrict__ enc_w,
    const float* __restrict__ enc_b, float* __restrict__ enc)
{
    __shared__ __align__(16) __bf16 Ms[CC * 3 * 68];     // [cc][dy][xx] ~26 KB
    __shared__ __align__(16) __bf16 Ws[112 * PITCH_K32]; // weight slice ~9 KB

    const int tid = threadIdx.x;
    const int n   = blockIdx.x >> 6;
    const int y   = blockIdx.x & 63;

    // Stage mid halo (rows y-1..y+1, cols -1..64) with zero padding.
    for (int i = tid; i < CC * 3 * 66; i += 128) {
        const int cc = i / 198, r = i % 198, dy = r / 66, xx = r % 66;
        const int yy = y + dy - 1, sx = xx - 1;
        __bf16 v = (__bf16)0.f;
        if ((unsigned)yy < (unsigned)HH && (unsigned)sx < (unsigned)WW)
            v = mid[((size_t)n * CC + cc) * HW + yy * WW + sx];
        Ms[(cc * 3 + dy) * 68 + xx] = v;
    }

    const int lane = tid & 31, wave = tid >> 5;
    const int col  = lane & 15, half = lane >> 4;
    const int xcol = wave * 16 + col;

    v8f acc[7] = {};
    for (int ks = 0; ks < 18; ++ks) {          // 576 / 32 K steps
        __syncthreads();                       // protect Ws reuse
        for (int i = tid; i < 112 * 32; i += 128) {
            const int row = i >> 5, kk = i & 31, k = kk + ks * 32;
            Ws[row * PITCH_K32 + kk] =
                (row < KC) ? (__bf16)enc_w[row * KD + k] : (__bf16)0.f;
        }
        __syncthreads();

        v16bf bfrag;                           // im2col gather from LDS
        #pragma unroll
        for (int v = 0; v < 8; ++v) {
            #pragma unroll
            for (int e = 0; e < 2; ++e) {
                const int kb = (v < 4 ? 2 * v : 16 + 2 * (v - 4)) + 8 * half + e;
                const int k  = ks * 32 + kb;
                const int cc = k / 9, rr = k % 9, dy = rr / 3, dx = rr % 3;
                bfrag[2 * v + e] = Ms[(cc * 3 + dy) * 68 + xcol + dx];
            }
        }
        #pragma unroll
        for (int m = 0; m < 7; ++m) {          // 112 padded output rows
            const v16bf afrag =
                load_frag16(&Ws[(m * 16 + col) * PITCH_K32], half);
            acc[m] = __builtin_amdgcn_wmma_f32_16x16x32_bf16(
                false, afrag, false, bfrag, (short)0, acc[m], false, false);
        }
    }

    #pragma unroll
    for (int m = 0; m < 7; ++m) {
        #pragma unroll
        for (int r = 0; r < 8; ++r) {
            const int row = m * 16 + r + 8 * half;
            if (row < KC)
                enc[(((size_t)n * KC + row) * HH + y) * WW + xcol] =
                    acc[m][r] + enc_b[row];
        }
    }
}

// ---------------------------------------------------------------------------
// Kernel 3: pixel-shuffle + softmax + content-aware reassembly.
// One block = 8x32 output tile (full-wave coalesced 128B stores).  The x halo
// for all 128 channels (6x18 src pixels each, ~54 KB) is staged in LDS via
// GLOBAL_LOAD_ASYNC_TO_LDS_B32 (padding elements zeroed with DS stores; the
// address sets are disjoint so the two counters need no cross-ordering).
// The 5x5 taps alias onto 3x3 source pixels after the /2 nearest-upsample
// mapping, so the 25 softmax weights fold (per parity) into 3x3 weights:
// 9 LDS loads + 9 FMAs per channel instead of 25.
// ---------------------------------------------------------------------------
__global__ __launch_bounds__(256) void carafe_reassemble(
    const float* __restrict__ x, const float* __restrict__ enc,
    float* __restrict__ out)
{
    __shared__ __align__(16) float xs[CIN * 6 * 18];  // [c][sy][sx] ~54 KB

    const int tid = threadIdx.x;
    int bid = blockIdx.x;
    const int n  = bid >> 6;  bid &= 63;       // 16*4 tiles per batch
    const int ty = bid >> 2, tx = bid & 3;
    const int y0 = ty * 8,  x0 = tx * 32;
    const int ys0 = (y0 >> 1) - 1, xs0 = (x0 >> 1) - 1;

    const float* xb = x + (size_t)n * CIN * HW;
    for (int i = tid; i < CIN * 6 * 18; i += 256) {
        const int c = i / 108, r = i % 108, ry = r / 18, rx = r % 18;
        const int sy = ys0 + ry, sx = xs0 + rx;
        const bool inb = (unsigned)sy < (unsigned)HH && (unsigned)sx < (unsigned)WW;
#if HAVE_ASYNC_LDS
        if (inb)
            __builtin_amdgcn_global_load_async_to_lds_b32(
                GPTR(xb + (c * HH + sy) * WW + sx), LPTR(&xs[i]), 0, 0);
        else
            xs[i] = 0.f;
#else
        xs[i] = inb ? xb[(c * HH + sy) * WW + sx] : 0.f;
#endif
    }
    wait_async_zero();
    __syncthreads();

    const int py = tid >> 5, px = tid & 31;
    const int yu = y0 + py, xu = x0 + px;

    // ---- softmax over 25 pixel-shuffled encoder channels ----
    const int sub = ((yu & 1) << 1) | (xu & 1);
    const float* ep = enc + (((size_t)n * KC + sub) * HH + (yu >> 1)) * WW + (xu >> 1);
    float w[25], mx = -1e30f;
    #pragma unroll
    for (int kk = 0; kk < 25; ++kk) { w[kk] = ep[kk * 4 * HW]; mx = fmaxf(mx, w[kk]); }
    float s = 0.f;
    #pragma unroll
    for (int kk = 0; kk < 25; ++kk) { w[kk] = __expf(w[kk] - mx); s += w[kk]; }
    const float inv = 1.f / s;
    #pragma unroll
    for (int kk = 0; kk < 25; ++kk) w[kk] *= inv;

    // ---- fold 5x5 weights to 3x3 effective weights (parity dependent) ----
    // tap i -> source offset: even parity {0,0,1,1,2}, odd parity {0,1,1,2,2}
    const bool pe = (py & 1) != 0, qe = (px & 1) != 0;
    float rowW[3][5];
    #pragma unroll
    for (int j = 0; j < 5; ++j) {
        const float w0 = w[j], w1 = w[5 + j], w2 = w[10 + j],
                    w3 = w[15 + j], w4 = w[20 + j];
        rowW[0][j] = pe ? w0        : (w0 + w1);
        rowW[1][j] = pe ? (w1 + w2) : (w2 + w3);
        rowW[2][j] = pe ? (w3 + w4) : w4;
    }
    float W3[3][3];
    #pragma unroll
    for (int a = 0; a < 3; ++a) {
        W3[a][0] = qe ? rowW[a][0]                : (rowW[a][0] + rowW[a][1]);
        W3[a][1] = qe ? (rowW[a][1] + rowW[a][2]) : (rowW[a][2] + rowW[a][3]);
        W3[a][2] = qe ? (rowW[a][3] + rowW[a][4]) : rowW[a][4];
    }

    // LDS coords of the 3x3 source window for this output pixel.
    const int lr = ((py - 2) >> 1) + 1;   // 0..3
    const int lc = ((px - 2) >> 1) + 1;   // 0..15

    float* op = out + (((size_t)n * CIN) * HU + yu) * WU + xu;
    #pragma unroll 4
    for (int c = 0; c < CIN; ++c) {
        const float* xc = &xs[c * 108];
        float acc = 0.f;
        #pragma unroll
        for (int a = 0; a < 3; ++a) {
            #pragma unroll
            for (int b = 0; b < 3; ++b)
                acc += W3[a][b] * xc[(lr + a) * 18 + (lc + b)];
        }
        op[(size_t)c * HU * WU] = acc;     // coalesced: 32 consecutive floats/wave
    }
}

// ---------------------------------------------------------------------------
extern "C" void kernel_launch(void* const* d_in, const int* in_sizes, int n_in,
                              void* d_out, int out_size, void* d_ws, size_t ws_size,
                              hipStream_t stream)
{
    (void)in_sizes; (void)n_in; (void)out_size; (void)ws_size;
    const float* x      = (const float*)d_in[0];
    const float* comp_w = (const float*)d_in[1];
    const float* comp_b = (const float*)d_in[2];
    const float* enc_w  = (const float*)d_in[3];
    const float* enc_b  = (const float*)d_in[4];
    float* out = (float*)d_out;

    // Workspace: mid (bf16, 2 MB) | enc (fp32, 6.25 MB)
    __bf16* mid = (__bf16*)d_ws;
    float*  enc = (float*)((char*)d_ws + (size_t)NB * CC * HW * sizeof(__bf16));

    carafe_compress_wmma<<<NB * (HW / 64), 128, 0, stream>>>(x, comp_w, comp_b, mid);
    carafe_encode_wmma  <<<NB * HH,        128, 0, stream>>>(mid, enc_w, enc_b, enc);
    carafe_reassemble   <<<NB * 16 * 4,    256, 0, stream>>>(x, enc, out);
}
